// Yolo5Segmentation_26706106646976
// MI455X (gfx1250) — matrix operand order repaired and checked by reading, not compiled, not run on hardware
//
#include <hip/hip_runtime.h>
#include <hip/hip_bf16.h>
#include <math.h>

typedef __attribute__((ext_vector_type(16))) _Float16 v16h;
typedef __attribute__((ext_vector_type(8)))  _Float16 v8h;
typedef __attribute__((ext_vector_type(8)))  float    v8f;

#define NANCH   25200
#define PSTRIDE 117
#define NCLS    80
#define NMASK   32
#define TOPK_N  1024
#define SORT_N  32768
#define MAXDET  300
#define MPAD    304          /* rows padded to 19*16 */
#define IMGSZ   640
#define PHW     160
#define PN      (PHW*PHW)    /* 25600 */
#define NTILE_M 19
#define NTILE_N (PN/16)      /* 1600 */
#define CONF_T  0.25f
#define IOU_T   0.45f
#define MAXWH   7680.0f

/* output layout (floats): masks_f, masks_bool, fb, fs, fcls */
#define OUT_MASKF 0UL
#define OUT_MASKB (300UL*640UL*640UL)
#define OUT_FB    (2UL*300UL*640UL*640UL)
#define OUT_FS    (OUT_FB + 1200UL)
#define OUT_FCLS  (OUT_FS + 300UL)

/* ---------------- Stage 1: decode anchors ---------------- */
__global__ void k_decode(const float* __restrict__ pred, float* __restrict__ scores,
                         float* __restrict__ boxes, int* __restrict__ cls) {
  int i = blockIdx.x * blockDim.x + threadIdx.x;
  if (i >= NANCH) return;
  const float* p = pred + (size_t)i * PSTRIDE;
  float obj = p[4];
  float best = -1e30f; int bj = 0;
  #pragma unroll 8
  for (int c = 0; c < NCLS; ++c) {
    float v = p[5 + c] * obj;
    if (v > best) { best = v; bj = c; }   /* strict > keeps first max (jnp.argmax) */
  }
  float cx = p[0], cy = p[1], w = p[2], h = p[3];
  boxes[i*4+0] = cx - 0.5f*w;
  boxes[i*4+1] = cy - 0.5f*h;
  boxes[i*4+2] = cx + 0.5f*w;
  boxes[i*4+3] = cy + 0.5f*h;
  bool valid = (obj > CONF_T) && (best > CONF_T);
  scores[i] = valid ? best : -1.0f;
  cls[i] = bj;
}

/* ---------------- Stage 2: exact top-k via bitonic sort of 64-bit keys ---- */
__global__ void k_sort_init(const float* __restrict__ scores,
                            unsigned long long* __restrict__ keys) {
  int i = blockIdx.x * blockDim.x + threadIdx.x;
  if (i >= SORT_N) return;
  if (i < NANCH) {
    unsigned u = __float_as_uint(scores[i]);
    u = (u & 0x80000000u) ? ~u : (u | 0x80000000u);  /* monotone float->uint */
    keys[i] = ((unsigned long long)u << 32) | (unsigned long long)(0xFFFFFFFFu - (unsigned)i);
  } else {
    keys[i] = 0ull;  /* -inf pad, sorts to the end of descending order */
  }
}

__global__ void k_sort_step(unsigned long long* __restrict__ keys, int j, int k) {
  int i = blockIdx.x * blockDim.x + threadIdx.x;
  if (i >= SORT_N) return;
  int ixj = i ^ j;
  if (ixj <= i) return;
  unsigned long long a = keys[i], b = keys[ixj];
  bool desc = ((i & k) == 0);              /* descending overall */
  if (desc ? (a < b) : (a > b)) { keys[i] = b; keys[ixj] = a; }
}

/* ---------------- Stage 3: gather top-1024 ---------------- */
__global__ void k_topk_gather(const unsigned long long* __restrict__ keys,
                              const float* __restrict__ scores,
                              const float* __restrict__ boxes,
                              const int* __restrict__ cls,
                              float* __restrict__ ts, float* __restrict__ tb,
                              int* __restrict__ tcls, float* __restrict__ tnms,
                              int* __restrict__ tidx) {
  int t = blockIdx.x * blockDim.x + threadIdx.x;
  if (t >= TOPK_N) return;
  unsigned long long k = keys[t];
  int idx = (int)(0xFFFFFFFFu - (unsigned)(k & 0xFFFFFFFFull));
  ts[t] = scores[idx];
  tcls[t] = cls[idx];
  tidx[t] = idx;
  float off = (float)cls[idx] * MAXWH;
  #pragma unroll
  for (int q = 0; q < 4; ++q) {
    float b = boxes[idx*4 + q];
    tb[t*4 + q] = b;
    tnms[t*4 + q] = b + off;
  }
}

/* ---------------- Stage 4: greedy NMS (single 1024-thread WG) ------------- */
__global__ void k_nms(const float* __restrict__ tnms, const float* __restrict__ ts,
                      int* __restrict__ keep) {
  __shared__ float bx[TOPK_N * 4];
  __shared__ int   keepSh[TOPK_N];
  __shared__ int   supFlag;
  int t = threadIdx.x;
  #pragma unroll
  for (int q = 0; q < 4; ++q) bx[t*4 + q] = tnms[t*4 + q];
  int validt = (ts[t] > CONF_T) ? 1 : 0;
  keepSh[t] = 0;
  __syncthreads();
  for (int i = 0; i < TOPK_N; ++i) {
    if (t == 0) supFlag = 0;
    __syncthreads();
    if (t < i && keepSh[t]) {
      float ax1 = bx[t*4+0], ay1 = bx[t*4+1], ax2 = bx[t*4+2], ay2 = bx[t*4+3];
      float bx1 = bx[i*4+0], by1 = bx[i*4+1], bx2 = bx[i*4+2], by2 = bx[i*4+3];
      float iw = fminf(ax2, bx2) - fmaxf(ax1, bx1); iw = fmaxf(iw, 0.0f);
      float ih = fminf(ay2, by2) - fmaxf(ay1, by1); ih = fmaxf(ih, 0.0f);
      float inter = iw * ih;
      float a1 = (ax2 - ax1) * (ay2 - ay1);
      float a2 = (bx2 - bx1) * (by2 - by1);
      float iou = inter / (a1 + a2 - inter + 1e-7f);
      if (iou > IOU_T) atomicOr(&supFlag, 1);
    }
    __syncthreads();
    if (t == i) keepSh[i] = (validt && !supFlag) ? 1 : 0;
    __syncthreads();
  }
  keep[t] = keepSh[t];
}

/* ---------------- Stage 5: compact kept -> finals, build f16 A ------------ */
__global__ void k_compact(const int* __restrict__ keep, const float* __restrict__ ts,
                          const float* __restrict__ tb, const int* __restrict__ tcls,
                          const int* __restrict__ tidx, const float* __restrict__ pred,
                          float* __restrict__ out, _Float16* __restrict__ Ah) {
  __shared__ int lst[MAXDET];
  __shared__ int cnt;
  if (threadIdx.x == 0) {
    int c = 0;
    for (int i = 0; i < TOPK_N && c < MAXDET; ++i)
      if (keep[i]) lst[c++] = i;
    cnt = c;
  }
  __syncthreads();
  for (int d = threadIdx.x; d < MAXDET; d += blockDim.x) {
    bool v = d < cnt;
    int src = v ? lst[d] : 0;
    out[OUT_FS   + d] = v ? ts[src] : 0.0f;              /* kept scores > CONF_T */
    out[OUT_FCLS + d] = v ? (float)tcls[src] : -1.0f;
    #pragma unroll
    for (int q = 0; q < 4; ++q)
      out[OUT_FB + (size_t)d*4 + q] = v ? tb[src*4 + q] : 0.0f;
    int pidx = v ? tidx[src] : 0;
    const float* cp = pred + (size_t)pidx * PSTRIDE + 5 + NCLS;
    for (int q = 0; q < NMASK; ++q)
      Ah[d*NMASK + q] = v ? (_Float16)cp[q] : (_Float16)0.0f;
  }
  for (int d = MAXDET + threadIdx.x; d < MPAD; d += blockDim.x)
    for (int q = 0; q < NMASK; ++q)
      Ah[d*NMASK + q] = (_Float16)0.0f;
}

/* ------------- Stage 6a: protos f32 -> f16, transposed to [n][k] ---------- */
__global__ void k_cvt_protos(const float* __restrict__ protos, _Float16* __restrict__ bt) {
  int i = blockIdx.x * blockDim.x + threadIdx.x;  /* i over 25600*32, coalesced writes */
  if (i >= NMASK * PN) return;
  int n = i >> 5;           /* proto pixel  */
  int k = i & 31;           /* channel      */
  bt[i] = (_Float16)protos[(size_t)k * PN + n];
}

/* ---------------- Stage 6b: WMMA GEMM (304x32 @ 32x25600) + sigmoid ------- */
/* One wave computes one 16x16 D tile. A/B use contiguous k order (k = 16*half + s).
   A row-major 304x32; B transposed [n][32] so each lane's K-run is contiguous. */
__global__ void k_mask_gemm(const _Float16* __restrict__ Ah,  /* 304 x 32 row-major */
                            const _Float16* __restrict__ Bt,  /* 25600 x 32 (n-major) */
                            float* __restrict__ M) {          /* 304 x 25600 (padded) */
  int wave = blockIdx.x * (blockDim.x >> 5) + (threadIdx.x >> 5);
  int lane = threadIdx.x & 31;
  int tm = wave / NTILE_N;
  int tn = wave - tm * NTILE_N;
  int lhalf = (lane < 16) ? 1 : 0;  /* 1 for lanes 0-15 */
  int l16   = lane & 15;

  /* A 16x32 f16: lanes 0-15 row=lane, K=0..15; lanes 16-31 K=16..31 */
  int arow = tm * 16 + l16;
  const _Float16* Ap = Ah + (size_t)arow * NMASK + (lhalf ? 0 : 16);
  v8h alo = *(const v8h*)(Ap);        /* 16B aligned contiguous */
  v8h ahi = *(const v8h*)(Ap + 8);
  v16h a;
  #pragma unroll
  for (int i = 0; i < 8; ++i) { a[i] = alo[i]; a[8 + i] = ahi[i]; }

  /* B 32x16 f16: lane = column n, lanes 0-15 hold K=0..15, lanes 16-31 K=16..31.
     Bt[n][k] => one contiguous 32B chunk per lane. */
  int bcol = tn * 16 + l16;
  int klo  = lhalf ? 0 : 16;
  v16h b = *(const v16h*)(Bt + (size_t)bcol * NMASK + klo);  /* 32B aligned */

  v8f c = {};
  c = __builtin_amdgcn_wmma_f32_16x16x32_f16(false, a, false, b,
                                             (short)0, c, false, false);

  /* D 16x16 f32: lane = column, VGPR v = row (rows 8..15 for lanes 16-31).
     Unconditional store into 304-row padded buffer (no exec divergence). */
  int r0 = tm * 16 + (lhalf ? 0 : 8);
  int n  = tn * 16 + l16;
  #pragma unroll
  for (int v = 0; v < 8; ++v) {
    float e = __expf(-c[v]);
    M[(size_t)(r0 + v) * PN + n] = __builtin_amdgcn_rcpf(1.0f + e);  /* sigmoid */
  }
}

/* ---------------- Stage 7: bilinear 160->640, crop, threshold ------------- */
__global__ void k_resize_crop(const float* __restrict__ M, const float* __restrict__ out_ro,
                              float* __restrict__ out) {
  int gid = blockIdx.x * blockDim.x + threadIdx.x;
  if (gid >= MAXDET * IMGSZ * IMGSZ) return;
  int d = gid / (IMGSZ * IMGSZ);
  int p = gid - d * (IMGSZ * IMGSZ);
  int y = p / IMGSZ;
  int x = p - y * IMGSZ;

  float fvalid = (out_ro[OUT_FS + d] > CONF_T) ? 1.0f : 0.0f;
  float x1 = out_ro[OUT_FB + (size_t)d*4 + 0];
  float y1 = out_ro[OUT_FB + (size_t)d*4 + 1];
  float x2 = out_ro[OUT_FB + (size_t)d*4 + 2];
  float y2 = out_ro[OUT_FB + (size_t)d*4 + 3];

  /* half-pixel bilinear, scale 4x, edge clamp */
  float sx = ((float)x + 0.5f) * 0.25f - 0.5f;
  float sy = ((float)y + 0.5f) * 0.25f - 0.5f;
  float fx0 = floorf(sx), fy0 = floorf(sy);
  float wx = sx - fx0,    wy = sy - fy0;
  int x0 = (int)fx0, y0 = (int)fy0;
  int x0c = min(max(x0,     0), PHW - 1);
  int x1c = min(max(x0 + 1, 0), PHW - 1);
  int y0c = min(max(y0,     0), PHW - 1);
  int y1c = min(max(y0 + 1, 0), PHW - 1);
  const float* md = M + (size_t)d * PN;
  __builtin_prefetch(md + y1c * PHW + x0c, 0, 3);  /* global_prefetch_b8 */
  float v00 = md[y0c*PHW + x0c], v01 = md[y0c*PHW + x1c];
  float v10 = md[y1c*PHW + x0c], v11 = md[y1c*PHW + x1c];
  float val = (1.0f - wy) * ((1.0f - wx) * v00 + wx * v01)
            +          wy * ((1.0f - wx) * v10 + wx * v11);

  float rr = (float)x, cc = (float)y;
  float inside = (rr >= x1 && rr < x2 && cc >= y1 && cc < y2) ? 1.0f : 0.0f;
  float mf = val * inside * fvalid;
  size_t o = (size_t)gid;
  out[OUT_MASKF + o] = mf;
  out[OUT_MASKB + o] = (mf > 0.5f) ? 1.0f : 0.0f;
}

/* ===================== host: orchestration ===================== */
extern "C" void kernel_launch(void* const* d_in, const int* in_sizes, int n_in,
                              void* d_out, int out_size, void* d_ws, size_t ws_size,
                              hipStream_t stream) {
  const float* pred   = (const float*)d_in[0];   /* (1,25200,117) */
  const float* protos = (const float*)d_in[1];   /* (32,160,160)  */
  float* out = (float*)d_out;

  /* workspace layout (256B aligned) */
  char* ws = (char*)d_ws;
  size_t off = 0;
  auto alloc = [&](size_t bytes) { size_t o = off; off += (bytes + 255) & ~(size_t)255; return o; };
  float*              scores = (float*)(ws + alloc(NANCH * 4));
  float*              boxes  = (float*)(ws + alloc(NANCH * 16));
  int*                cls    = (int*)  (ws + alloc(NANCH * 4));
  unsigned long long* keys   = (unsigned long long*)(ws + alloc(SORT_N * 8));
  float*              ts     = (float*)(ws + alloc(TOPK_N * 4));
  float*              tb     = (float*)(ws + alloc(TOPK_N * 16));
  int*                tcls   = (int*)  (ws + alloc(TOPK_N * 4));
  float*              tnms   = (float*)(ws + alloc(TOPK_N * 16));
  int*                tidx   = (int*)  (ws + alloc(TOPK_N * 4));
  int*                keep   = (int*)  (ws + alloc(TOPK_N * 4));
  _Float16*           Ah     = (_Float16*)(ws + alloc((size_t)MPAD * NMASK * 2));
  _Float16*           Bt     = (_Float16*)(ws + alloc((size_t)PN * NMASK * 2));
  float*              Mbuf   = (float*)(ws + alloc((size_t)MPAD * PN * 4));
  (void)ws_size; (void)in_sizes; (void)n_in; (void)out_size;

  k_decode<<<(NANCH + 255) / 256, 256, 0, stream>>>(pred, scores, boxes, cls);

  k_sort_init<<<SORT_N / 256, 256, 0, stream>>>(scores, keys);
  for (int k = 2; k <= SORT_N; k <<= 1)
    for (int j = k >> 1; j > 0; j >>= 1)
      k_sort_step<<<SORT_N / 256, 256, 0, stream>>>(keys, j, k);

  k_topk_gather<<<TOPK_N / 256, 256, 0, stream>>>(keys, scores, boxes, cls,
                                                  ts, tb, tcls, tnms, tidx);

  k_nms<<<1, TOPK_N, 0, stream>>>(tnms, ts, keep);

  k_compact<<<1, 512, 0, stream>>>(keep, ts, tb, tcls, tidx, pred, out, Ah);

  k_cvt_protos<<<(NMASK * PN + 255) / 256, 256, 0, stream>>>(protos, Bt);

  /* 19*1600 = 30400 waves, 8 waves (256 threads) per block -> 3800 blocks */
  k_mask_gemm<<<(NTILE_M * NTILE_N) / 8, 256, 0, stream>>>(Ah, Bt, Mbuf);

  k_resize_crop<<<(MAXDET * IMGSZ * IMGSZ + 255) / 256, 256, 0, stream>>>(Mbuf, out, out);
}
